// MGMaskAttention_22771916603917
// MI455X (gfx1250) — matrix operand order repaired and checked by reading, not compiled, not run on hardware
//
#include <hip/hip_runtime.h>
#include <hip/hip_bf16.h>
#include <math.h>

// Problem constants
#define TSEQ  2048
#define NEMBD 1024
#define NHEAD 16
#define DHEAD 64
#define BATCH 4

typedef __attribute__((ext_vector_type(16))) __bf16 bf16x16;
typedef __attribute__((ext_vector_type(8)))  float  f32x8;

union FragAB { uint4 q[2]; bf16x16 v; };

// CDNA5 async global->LDS copy path (guarded; falls back to VGPR staging)
#if defined(__HIP_DEVICE_COMPILE__) && __has_builtin(__builtin_amdgcn_global_load_async_to_lds_b128)
#define HAVE_ASYNC_LDS 1
typedef int v4i_t __attribute__((vector_size(16)));
typedef __attribute__((address_space(1))) v4i_t gv4i;   // global int4
typedef __attribute__((address_space(3))) v4i_t lv4i;   // LDS int4
#define ASYNC_WAIT() asm volatile("s_wait_asynccnt 0" ::: "memory")
#endif

__device__ __forceinline__ f32x8 f32x8_zero() {
    f32x8 z = {0.f,0.f,0.f,0.f,0.f,0.f,0.f,0.f};
    return z;
}

__device__ __forceinline__ f32x8 wmma_bf16(bf16x16 a, bf16x16 b, f32x8 c) {
    // D = A(16x32 bf16) x B(32x16 bf16) + C(16x16 f32)
    return __builtin_amdgcn_wmma_f32_16x16x32_bf16(
        /*neg_a=*/false, a, /*neg_b=*/false, b,
        /*c_mod=*/(short)0, c, /*reuse_a=*/false, /*reuse_b=*/false);
}

// ---------------------------------------------------------------------------
// Elementwise f32 -> bf16 conversion
// ---------------------------------------------------------------------------
__global__ void cvt_f32_bf16_kernel(const float* __restrict__ in,
                                    __bf16* __restrict__ out, int n) {
    int i = blockIdx.x * blockDim.x + threadIdx.x;
    if (i < n) out[i] = (__bf16)in[i];
}

// Transpose + convert: in [K][N] f32 row-major -> out [N][K] bf16
__global__ void transpose_f32_bf16_kernel(const float* __restrict__ in,
                                          __bf16* __restrict__ out,
                                          int K, int N) {
    int i = blockIdx.x * blockDim.x + threadIdx.x;
    if (i < K * N) {
        int k = i / N;
        int n = i - k * N;
        out[(size_t)n * K + k] = (__bf16)in[i];
    }
}

// Per-K-step fragment loads + 8 WMMAs for one wave (32x64 sub-tile).
__device__ __forceinline__ void gemm_compute_step(
    const __bf16 (*__restrict__ As)[40], const __bf16 (*__restrict__ Bs)[40],
    int wM, int wN, int ln, int half, f32x8 acc[2][4])
{
    FragAB af[2];
    for (int mt = 0; mt < 2; ++mt) {
        const __bf16* ar = &As[wM + mt * 16 + ln][0];
        af[mt].q[0] = *(const uint4*)(ar + half * 16);      // K = h*16..+7
        af[mt].q[1] = *(const uint4*)(ar + half * 16 + 8);  // K = h*16+8..+15
    }
    FragAB bfr[4];
    for (int nt = 0; nt < 4; ++nt) {
        const __bf16* br = &Bs[wN + nt * 16 + ln][half * 16]; // K = h*16..+15
        bfr[nt].q[0] = *(const uint4*)(br);
        bfr[nt].q[1] = *(const uint4*)(br + 8);
    }
    for (int mt = 0; mt < 2; ++mt)
        for (int nt = 0; nt < 4; ++nt)
            acc[mt][nt] = wmma_bf16(af[mt].v, bfr[nt].v, acc[mt][nt]);
}

// ---------------------------------------------------------------------------
// Tiled bf16 WMMA GEMM:  C[M][N] = A[M][K] * Bt[N][K]^T + bias
// Block tile 128x128, K-step 32, 8 waves (4 along M x 2 along N),
// each wave owns a 32x64 sub-tile = 2x4 accumulators of 16x16.
// Staging: async global->LDS (double buffered) when available.
// mode 0: QKV epilogue -> scatter into Q[BH][T][D] (scaled), K[BH][T][D],
//         Vt[BH][D][T], all bf16.
// mode 1: plain f32 output with bias (projection).
// ---------------------------------------------------------------------------
__global__ __launch_bounds__(256) void gemm_bf16_wmma_kernel(
    const __bf16* __restrict__ A,    // [M][K]
    const __bf16* __restrict__ Bt,   // [N][K]
    const float*  __restrict__ bias, // [N]
    float*        __restrict__ outF, // mode 1
    __bf16*       __restrict__ outQ, // mode 0
    __bf16*       __restrict__ outK, // mode 0
    __bf16*       __restrict__ outV, // mode 0 (transposed V)
    int M, int N, int K, int mode)
{
    __shared__ __align__(16) __bf16 As[2][128][40]; // 80B row stride, 16B aligned
    __shared__ __align__(16) __bf16 Bs[2][128][40];

    const int tid  = threadIdx.x;
    const int lane = tid & 31;
    const int half = lane >> 4;   // lane-half selects K/M sub-blocks
    const int ln   = lane & 15;
    const int wid  = tid >> 5;
    const int wM   = (wid >> 1) * 32; // wave M offset in block tile
    const int wN   = (wid & 1) * 64;  // wave N offset in block tile

    const int n0 = blockIdx.x * 128;
    const int m0 = blockIdx.y * 128;

    f32x8 acc[2][4];
    for (int mt = 0; mt < 2; ++mt)
        for (int nt = 0; nt < 4; ++nt)
            acc[mt][nt] = f32x8_zero();

    const int srow = tid >> 1;           // each thread stages one 16B chunk
    const int skh  = (tid & 1) * 16;     // of A and one of B per K-step

#ifdef HAVE_ASYNC_LDS
    // ---- double-buffered async global->LDS pipeline (ASYNCcnt tracked) ----
    const int nsteps = K >> 5;
    {
        __builtin_amdgcn_global_load_async_to_lds_b128(
            (gv4i*)(void*)(A + (size_t)(m0 + srow) * K + skh),
            (lv4i*)(void*)&As[0][srow][skh], 0, 0);
        __builtin_amdgcn_global_load_async_to_lds_b128(
            (gv4i*)(void*)(Bt + (size_t)(n0 + srow) * K + skh),
            (lv4i*)(void*)&Bs[0][srow][skh], 0, 0);
        ASYNC_WAIT();
        __syncthreads();
    }
    for (int i = 0; i < nsteps; ++i) {
        const int cur = i & 1;
        if (i + 1 < nsteps) {
            const int nxt = cur ^ 1;
            const int ks  = (i + 1) << 5;
            __builtin_amdgcn_global_load_async_to_lds_b128(
                (gv4i*)(void*)(A + (size_t)(m0 + srow) * K + ks + skh),
                (lv4i*)(void*)&As[nxt][srow][skh], 0, 0);
            __builtin_amdgcn_global_load_async_to_lds_b128(
                (gv4i*)(void*)(Bt + (size_t)(n0 + srow) * K + ks + skh),
                (lv4i*)(void*)&Bs[nxt][srow][skh], 0, 0);
        }
        gemm_compute_step(As[cur], Bs[cur], wM, wN, ln, half, acc);
        ASYNC_WAIT();        // my async writes into the next buffer landed
        __syncthreads();     // everyone done reading cur + writes visible
    }
#else
    // ---- fallback: VGPR staging, single buffer ----
    for (int ks = 0; ks < K; ks += 32) {
        *(uint4*)&As[0][srow][skh] =
            *(const uint4*)&A[(size_t)(m0 + srow) * K + ks + skh];
        *(uint4*)&Bs[0][srow][skh] =
            *(const uint4*)&Bt[(size_t)(n0 + srow) * K + ks + skh];
        __syncthreads();
        gemm_compute_step(As[0], Bs[0], wM, wN, ln, half, acc);
        __syncthreads();
    }
#endif

    // ---- epilogue ----
    const int mBase = m0 + wM;
    for (int mt = 0; mt < 2; ++mt) {
        for (int nt = 0; nt < 4; ++nt) {
            const int gn = n0 + wN + nt * 16 + ln;
            const float bv = bias[gn];
            f32x8 a = acc[mt][nt];
            if (mode == 1) {
                for (int r = 0; r < 8; ++r) {
                    int gm = mBase + mt * 16 + r + 8 * half;
                    outF[(size_t)gm * N + gn] = a[r] + bv;
                }
            } else {
                const int part = gn >> 10;       // 0:Q 1:K 2:V
                const int cc   = gn & 1023;
                const int hh   = cc >> 6;        // head
                const int dd   = cc & 63;        // head dim
                for (int r = 0; r < 8; ++r) {
                    int gm = mBase + mt * 16 + r + 8 * half;
                    int bI = gm >> 11;           // batch (T=2048)
                    int tt = gm & 2047;          // seq pos
                    int bh = bI * NHEAD + hh;
                    float v = a[r] + bv;
                    if (part == 0) {             // Q, fold in 1/sqrt(D)
                        outQ[((size_t)bh * TSEQ + tt) * DHEAD + dd] =
                            (__bf16)(v * 0.125f);
                    } else if (part == 1) {      // K
                        outK[((size_t)bh * TSEQ + tt) * DHEAD + dd] = (__bf16)v;
                    } else {                     // V transposed: [bh][d][t]
                        outV[((size_t)bh * DHEAD + dd) * TSEQ + tt] = (__bf16)v;
                    }
                }
            }
        }
    }
}

// ---------------------------------------------------------------------------
// Flash attention (causal), bf16 WMMA, f32 online softmax.
// grid = (T/64, B*H), block = 128 (4 waves). Each wave: 16 queries.
// Q/K: [BH][T][64] bf16 (Q pre-scaled by 1/sqrt(D)); Vt: [BH][64][T] bf16.
// Output Y: [B][T][C] bf16 (feeds projection GEMM directly).
// ---------------------------------------------------------------------------
__global__ __launch_bounds__(128) void attn_flash_wmma_kernel(
    const __bf16* __restrict__ Q,
    const __bf16* __restrict__ Kmat,
    const __bf16* __restrict__ Vt,
    __bf16* __restrict__ Y)
{
    __shared__ __align__(16) __bf16 Plds[4][16][40]; // per-wave P staging

    const int lane = threadIdx.x & 31;
    const int half = lane >> 4;
    const int ln   = lane & 15;
    const int wid  = threadIdx.x >> 5;

    const int bh = blockIdx.y;
    const int b  = bh >> 4;
    const int h  = bh & 15;
    const int q0 = blockIdx.x * 64 + wid * 16;

    const __bf16* Qp  = Q    + (size_t)bh * TSEQ * DHEAD;
    const __bf16* Kp  = Kmat + (size_t)bh * TSEQ * DHEAD;
    const __bf16* Vtp = Vt   + (size_t)bh * DHEAD * TSEQ;

    // Persistent Q fragments: 16 queries x 64 dims = 2 A-frags (K=32 each)
    FragAB qf[2];
    {
        const __bf16* qrow = Qp + (size_t)(q0 + ln) * DHEAD;
        for (int c = 0; c < 2; ++c) {
            qf[c].q[0] = *(const uint4*)(qrow + c * 32 + half * 16);
            qf[c].q[1] = *(const uint4*)(qrow + c * 32 + half * 16 + 8);
        }
    }

    float m_i[8], l_i[8];
    for (int r = 0; r < 8; ++r) { m_i[r] = -INFINITY; l_i[r] = 0.f; }
    f32x8 acc[4];
    for (int dt = 0; dt < 4; ++dt) acc[dt] = f32x8_zero();

    for (int kb = 0; kb < q0 + 16; kb += 32) {
        // ---- prefetch next key/value block (global_prefetch_b8) ----
        if (kb + 32 < q0 + 16) {
            __builtin_prefetch(Kp + (size_t)(kb + 32 + lane) * DHEAD, 0, 0);
            __builtin_prefetch(Vtp + (size_t)(2 * lane) * TSEQ + kb + 32, 0, 0);
            __builtin_prefetch(Vtp + (size_t)(2 * lane + 1) * TSEQ + kb + 32, 0, 0);
        }

        // ---- S = Q Kt : two 16x16 tiles over 32 keys ----
        FragAB kf[2][2];
        for (int nt = 0; nt < 2; ++nt) {
            const __bf16* krow = Kp + (size_t)(kb + nt * 16 + ln) * DHEAD;
            for (int c = 0; c < 2; ++c) {
                kf[nt][c].q[0] = *(const uint4*)(krow + c * 32 + half * 16);
                kf[nt][c].q[1] = *(const uint4*)(krow + c * 32 + half * 16 + 8);
            }
        }
        f32x8 s[2];
        s[0] = f32x8_zero();
        s[1] = f32x8_zero();
        for (int nt = 0; nt < 2; ++nt)
            for (int c = 0; c < 2; ++c)
                s[nt] = wmma_bf16(qf[c].v, kf[nt][c].v, s[nt]);

        // ---- causal mask (only on the diagonal iteration; uniform branch) ----
        if (kb + 32 > q0) {
            for (int nt = 0; nt < 2; ++nt) {
                int key = kb + nt * 16 + ln;
                for (int r = 0; r < 8; ++r) {
                    int qq = q0 + r + 8 * half;
                    if (key > qq) s[nt][r] = -INFINITY;
                }
            }
        }

        // ---- online softmax: row stats via 16-lane xor reductions ----
        float mnew[8], alpha[8];
        for (int r = 0; r < 8; ++r) {
            float rm = fmaxf(s[0][r], s[1][r]);
            for (int off = 1; off < 16; off <<= 1)
                rm = fmaxf(rm, __shfl_xor(rm, off, 16));
            mnew[r]  = fmaxf(m_i[r], rm);
            alpha[r] = __expf(m_i[r] - mnew[r]);
            m_i[r]   = mnew[r];
        }
        for (int nt = 0; nt < 2; ++nt)
            for (int r = 0; r < 8; ++r)
                s[nt][r] = __expf(s[nt][r] - mnew[r]); // masked -> 0
        for (int r = 0; r < 8; ++r) {
            float rs = s[0][r] + s[1][r];
            for (int off = 1; off < 16; off <<= 1)
                rs += __shfl_xor(rs, off, 16);
            l_i[r] = l_i[r] * alpha[r] + rs;
        }
        for (int dt = 0; dt < 4; ++dt)
            for (int r = 0; r < 8; ++r)
                acc[dt][r] *= alpha[r];

        // ---- P: C-layout f32 -> bf16 A-layout via per-wave LDS ----
        for (int nt = 0; nt < 2; ++nt)
            for (int r = 0; r < 8; ++r)
                Plds[wid][r + 8 * half][nt * 16 + ln] = (__bf16)s[nt][r];

        FragAB pf;
        {
            const __bf16* prow = &Plds[wid][ln][0];
            pf.q[0] = *(const uint4*)(prow + half * 16);
            pf.q[1] = *(const uint4*)(prow + half * 16 + 8);
        }

        // ---- O += P V : Vt[d][t] is contiguous over keys ----
        for (int dt = 0; dt < 4; ++dt) {
            const __bf16* vrow = Vtp + (size_t)(dt * 16 + ln) * TSEQ + kb;
            FragAB vf;
            vf.q[0] = *(const uint4*)(vrow + half * 16);
            vf.q[1] = *(const uint4*)(vrow + half * 16 + 8);
            acc[dt] = wmma_bf16(pf.v, vf.v, acc[dt]);
        }
    }

    // ---- normalize and write Y[b][t][h*64+d] as bf16 ----
    float invl[8];
    for (int r = 0; r < 8; ++r) invl[r] = 1.0f / l_i[r];
    for (int dt = 0; dt < 4; ++dt) {
        for (int r = 0; r < 8; ++r) {
            int t = q0 + r + 8 * half;
            size_t idx = ((size_t)(b * TSEQ + t)) * NEMBD + h * DHEAD + dt * 16 + ln;
            Y[idx] = (__bf16)(acc[dt][r] * invl[r]);
        }
    }
}

// ---------------------------------------------------------------------------
// Host launcher
// ---------------------------------------------------------------------------
extern "C" void kernel_launch(void* const* d_in, const int* in_sizes, int n_in,
                              void* d_out, int out_size, void* d_ws, size_t ws_size,
                              hipStream_t stream) {
    const float* x     = (const float*)d_in[0]; // [4,2048,1024]
    const float* Wqkv  = (const float*)d_in[1]; // [1024,3072]
    const float* bqkv  = (const float*)d_in[2]; // [3072]
    const float* Wproj = (const float*)d_in[3]; // [1024,1024]
    const float* bproj = (const float*)d_in[4]; // [1024]
    float* out = (float*)d_out;                 // [4,2048,1024]

    const size_t M  = (size_t)BATCH * TSEQ;     // 8192
    const size_t BH = (size_t)BATCH * NHEAD;    // 64

    char* p = (char*)d_ws;
    auto carve = [&](size_t bytes) -> char* {
        char* r = p;
        p += (bytes + 255) & ~(size_t)255;
        return r;
    };
    __bf16* xb     = (__bf16*)carve(M * NEMBD * 2);                 // x in bf16
    __bf16* wqkvT  = (__bf16*)carve((size_t)3 * NEMBD * NEMBD * 2); // [3C][C]
    __bf16* wprojT = (__bf16*)carve((size_t)NEMBD * NEMBD * 2);     // [C][C]
    __bf16* Qb     = (__bf16*)carve(BH * TSEQ * DHEAD * 2);
    __bf16* Kb     = (__bf16*)carve(BH * TSEQ * DHEAD * 2);
    __bf16* Vtb    = (__bf16*)carve(BH * DHEAD * TSEQ * 2);
    __bf16* Yb     = (__bf16*)carve(M * NEMBD * 2);

    // 1) precision conversion / weight transposes
    {
        int n = (int)(M * NEMBD);
        cvt_f32_bf16_kernel<<<(n + 255) / 256, 256, 0, stream>>>(x, xb, n);
    }
    {
        int n = NEMBD * 3 * NEMBD;
        transpose_f32_bf16_kernel<<<(n + 255) / 256, 256, 0, stream>>>(
            Wqkv, wqkvT, NEMBD, 3 * NEMBD);
    }
    {
        int n = NEMBD * NEMBD;
        transpose_f32_bf16_kernel<<<(n + 255) / 256, 256, 0, stream>>>(
            Wproj, wprojT, NEMBD, NEMBD);
    }

    // 2) fused QKV GEMM (+bias, +1/sqrt(D) on Q, reshape, V transposed)
    gemm_bf16_wmma_kernel<<<dim3(3 * NEMBD / 128, M / 128), 256, 0, stream>>>(
        xb, wqkvT, bqkv, nullptr, Qb, Kb, Vtb,
        (int)M, 3 * NEMBD, NEMBD, /*mode=*/0);

    // 3) causal flash attention
    attn_flash_wmma_kernel<<<dim3(TSEQ / 64, (unsigned)BH), 128, 0, stream>>>(
        Qb, Kb, Vtb, Yb);

    // 4) output projection GEMM (+bias) -> f32 d_out
    gemm_bf16_wmma_kernel<<<dim3(NEMBD / 128, M / 128), 256, 0, stream>>>(
        Yb, wprojT, bproj, out, nullptr, nullptr, nullptr,
        (int)M, NEMBD, NEMBD, /*mode=*/1);
}
